// SCAttEnc_7146825581190
// MI455X (gfx1250) — matrix operand order repaired and checked by reading, not compiled, not run on hardware
//
#include <hip/hip_runtime.h>

// Problem constants (match reference)
constexpr int kB   = 64;
constexpr int kH   = 8;
constexpr int kM   = 1024;
constexpr int kC   = 128;
constexpr int kMID = 64;

typedef __attribute__((ext_vector_type(16))) __bf16 v16bf;
typedef __attribute__((ext_vector_type(8)))  float  v8f;

union Frag16 { v16bf v; unsigned u[8]; };

// Pack two fp32 -> one dword of two bf16 (RNE), lo first.
__device__ __forceinline__ unsigned pack_bf2(float lo, float hi) {
#if __has_builtin(__builtin_amdgcn_cvt_pk_bf16_f32)
  auto r = __builtin_amdgcn_cvt_pk_bf16_f32(lo, hi);
  return __builtin_bit_cast(unsigned, r);
#else
  unsigned a = __float_as_uint(lo);
  unsigned b = __float_as_uint(hi);
  a = a + 0x7FFFu + ((a >> 16) & 1u);
  b = b + 0x7FFFu + ((b >> 16) & 1u);
  return (a >> 16) | (b & 0xFFFF0000u);
#endif
}

__global__ __launch_bounds__(256) void scatt_kernel(
    const float* __restrict__ query,   const float* __restrict__ key_feat,
    const int*   __restrict__ att_mask,const float* __restrict__ value1,
    const float* __restrict__ value2,  const float* __restrict__ w_basic,
    const float* __restrict__ b_basic, const float* __restrict__ w_last,
    const float* __restrict__ b_last,  const float* __restrict__ w_last2,
    const float* __restrict__ b_last2, float* __restrict__ out)
{
  __shared__ __align__(16) float q_lds[kC];
  __shared__ unsigned wbp[(kC / 2) * kMID];   // w_basic, bf16 pairs, B-frag layout
  __shared__ float bb_lds[kMID];
  __shared__ float wl_lds[kMID];
  __shared__ float maskf[kM];
  __shared__ float alpha[kM];
  __shared__ float pooled[kMID];
  __shared__ float red[256];

  const int tid = threadIdx.x;
  const int bh  = blockIdx.x;          // b*H + h
  const int b   = bh >> 3;
  const int h   = bh & 7;

  const float* qg   = query    + (size_t)bh * kC;
  const float* kfg  = key_feat + (size_t)bh * kM * kC;
  const int*   amg  = att_mask + (size_t)b  * kM;
  const float* v1g  = value1   + (size_t)bh * kC;
  const float* v2g  = value2   + (size_t)bh * kM * kC;
  const float* wbg  = w_basic  + (size_t)h  * kC * kMID;
  const float* wl2g = w_last2  + (size_t)h  * kMID * kC;

  // ---------------- preload ----------------
  if (tid < kC)  q_lds[tid] = qg[tid];
  if (tid < kMID) {
    bb_lds[tid] = b_basic[h * kMID + tid];
    wl_lds[tid] = w_last[h * kMID + tid];
    pooled[tid] = 0.f;
  }
  for (int i = tid; i < (kC / 2) * kMID; i += 256) {
    int cp = i >> 6;      // c-pair index 0..63
    int o  = i & 63;      // mid index
    wbp[i] = pack_bf2(wbg[(2 * cp) * kMID + o], wbg[(2 * cp + 1) * kMID + o]);
  }
  for (int i = tid; i < kM; i += 256) maskf[i] = (float)amg[i];
  const float blast = b_last[h];
  __syncthreads();

  // ---------------- phase 1: WMMA GEMM over M tiles ----------------
  const int wave = tid >> 5;
  const int lane = tid & 31;
  const int ln   = lane & 15;
  const int hi   = lane >> 4;

  // Hoist all 16 B fragments (kt x nt) into registers: loop-invariant over M.
  // Layout: lane = N column (ln), VGPR j holds K = hi*8 + 2j + (j>=4 ? 8 : 0), +1.
  Frag16 Bfr[16];
  #pragma unroll
  for (int kt = 0; kt < 4; ++kt)
    #pragma unroll
    for (int nt = 0; nt < 4; ++nt)
      #pragma unroll
      for (int j = 0; j < 8; ++j)
        Bfr[kt * 4 + nt].u[j] = wbp[(kt * 16 + hi * 4 + j + (j & 4)) * kMID + nt * 16 + ln];

  float poolacc[4] = {0.f, 0.f, 0.f, 0.f};

  for (int t = wave; t < kM / 16; t += 8) {
    v8f acc[4];
    #pragma unroll
    for (int nt = 0; nt < 4; ++nt)
      #pragma unroll
      for (int i = 0; i < 8; ++i) acc[nt][i] = 0.f;

    const float* krow = kfg + (size_t)(t * 16 + ln) * kC;   // A row m = ln

    #pragma unroll
    for (int kt = 0; kt < 4; ++kt) {
      // A fragment: lanes 0-15 K in {0..7,16..23}, lanes 16-31 K in {8..15,24..31}
      const int c0 = kt * 32 + hi * 8;
      const int c1 = c0 + 16;
      float4 ka0 = *reinterpret_cast<const float4*>(krow + c0);
      float4 ka1 = *reinterpret_cast<const float4*>(krow + c0 + 4);
      float4 kb0 = *reinterpret_cast<const float4*>(krow + c1);
      float4 kb1 = *reinterpret_cast<const float4*>(krow + c1 + 4);
      float4 qa0 = *reinterpret_cast<const float4*>(q_lds + c0);
      float4 qa1 = *reinterpret_cast<const float4*>(q_lds + c0 + 4);
      float4 qb0 = *reinterpret_cast<const float4*>(q_lds + c1);
      float4 qb1 = *reinterpret_cast<const float4*>(q_lds + c1 + 4);
      Frag16 A;
      A.u[0] = pack_bf2(qa0.x * ka0.x, qa0.y * ka0.y);
      A.u[1] = pack_bf2(qa0.z * ka0.z, qa0.w * ka0.w);
      A.u[2] = pack_bf2(qa1.x * ka1.x, qa1.y * ka1.y);
      A.u[3] = pack_bf2(qa1.z * ka1.z, qa1.w * ka1.w);
      A.u[4] = pack_bf2(qb0.x * kb0.x, qb0.y * kb0.y);
      A.u[5] = pack_bf2(qb0.z * kb0.z, qb0.w * kb0.w);
      A.u[6] = pack_bf2(qb1.x * kb1.x, qb1.y * kb1.y);
      A.u[7] = pack_bf2(qb1.z * kb1.z, qb1.w * kb1.w);

      #pragma unroll
      for (int nt = 0; nt < 4; ++nt) {
        acc[nt] = __builtin_amdgcn_wmma_f32_16x16x32_bf16(
            false, A.v, false, Bfr[kt * 4 + nt].v, (short)0, acc[nt],
            false, false);
      }
    }

    // fused epilogue: bias + relu, masked pool partials, w_last dot
    float alphapart[8];
    #pragma unroll
    for (int v = 0; v < 8; ++v) alphapart[v] = 0.f;
    #pragma unroll
    for (int nt = 0; nt < 4; ++nt) {
      const int   o    = nt * 16 + ln;
      const float bias = bb_lds[o];
      const float wlo  = wl_lds[o];
      #pragma unroll
      for (int v = 0; v < 8; ++v) {
        float x = fmaxf(acc[nt][v] + bias, 0.f);       // D row = v + 8*hi, col = o
        poolacc[nt]  += x * maskf[t * 16 + v + hi * 8];
        alphapart[v] += x * wlo;
      }
    }
    #pragma unroll
    for (int v = 0; v < 8; ++v) {      // reduce over 16 N-lanes of same half
      float s = alphapart[v];
      s += __shfl_xor(s, 1);
      s += __shfl_xor(s, 2);
      s += __shfl_xor(s, 4);
      s += __shfl_xor(s, 8);
      if (ln == v) alpha[t * 16 + v + hi * 8] = s + blast;
    }
  }
  #pragma unroll
  for (int nt = 0; nt < 4; ++nt)
    atomicAdd(&pooled[nt * 16 + ln], poolacc[nt]);   // ds_add_f32
  __syncthreads();

  // ---------------- phase 2: masked softmax + streams ----------------
  float lm = -1e30f;
  for (int m = tid; m < kM; m += 256)
    lm = fmaxf(lm, maskf[m] != 0.f ? alpha[m] : -1e9f);
  red[tid] = lm; __syncthreads();
  for (int s = 128; s > 0; s >>= 1) { if (tid < s) red[tid] = fmaxf(red[tid], red[tid + s]); __syncthreads(); }
  const float mx = red[0];
  __syncthreads();

  float ls = 0.f;
  for (int m = tid; m < kM; m += 256) {
    float a = maskf[m] != 0.f ? __expf(alpha[m] - mx) : 0.f;
    alpha[m] = a;
    ls += a;
  }
  red[tid] = ls; __syncthreads();
  for (int s = 128; s > 0; s >>= 1) { if (tid < s) red[tid] += red[tid + s]; __syncthreads(); }
  const float denom = red[0];
  __syncthreads();

  float lc = 0.f;
  for (int m = tid; m < kM; m += 256) lc += maskf[m];
  red[tid] = lc; __syncthreads();
  for (int s = 128; s > 0; s >>= 1) { if (tid < s) red[tid] += red[tid + s]; __syncthreads(); }
  const float cnt = red[0];
  __syncthreads();

  if (tid < kMID) pooled[tid] = pooled[tid] / cnt;   // masked mean

  // v2 = softmax . value2 : threads 0..127 = channel, two M-halves in parallel
  const int c    = tid & 127;
  const int half = tid >> 7;
  const float* v2p = v2g + (size_t)half * (kM / 2) * kC;
  float vacc = 0.f;
  for (int m = 0; m < kM / 2; ++m)
    vacc += alpha[half * (kM / 2) + m] * v2p[(size_t)m * kC + c];
  red[tid] = vacc; __syncthreads();

  if (tid < kC) {
    float v2v = (red[tid] + red[tid + 128]) / denom;
    float a = b_last2[h * kC + c];
    #pragma unroll 4
    for (int o = 0; o < kMID; ++o) a += pooled[o] * wl2g[o * kC + c];
    float sig = 1.f / (1.f + __expf(-a));
    out[(size_t)bh * kC + c] = v1g[c] * v2v * sig;
  }
}

extern "C" void kernel_launch(void* const* d_in, const int* in_sizes, int n_in,
                              void* d_out, int out_size, void* d_ws, size_t ws_size,
                              hipStream_t stream) {
  (void)in_sizes; (void)n_in; (void)out_size; (void)d_ws; (void)ws_size;
  const float* query    = (const float*)d_in[0];
  const float* key_feat = (const float*)d_in[1];
  const int*   att_mask = (const int*)  d_in[2];
  const float* value1   = (const float*)d_in[3];
  const float* value2   = (const float*)d_in[4];
  const float* w_basic  = (const float*)d_in[5];
  const float* b_basic  = (const float*)d_in[6];
  const float* w_last   = (const float*)d_in[7];
  const float* b_last   = (const float*)d_in[8];
  const float* w_last2  = (const float*)d_in[9];
  const float* b_last2  = (const float*)d_in[10];
  float* out = (float*)d_out;

  dim3 grid(kB * kH);   // 512 workgroups, one per (b,h)
  dim3 block(256);      // 8 wave32
  scatt_kernel<<<grid, block, 0, stream>>>(query, key_feat, att_mask, value1,
                                           value2, w_basic, b_basic, w_last,
                                           b_last, w_last2, b_last2, out);
}
